// Fi_GNN_32615981646072
// MI455X (gfx1250) — matrix operand order repaired and checked by reading, not compiled, not run on hardware
//
#include <hip/hip_runtime.h>
#include <cstdint>
#include <cstddef>

// ---------------- types ----------------
typedef __bf16 bf16;
typedef __attribute__((ext_vector_type(2)))  __bf16 v2bf;
typedef __attribute__((ext_vector_type(16))) __bf16 v16bf;
typedef __attribute__((ext_vector_type(8)))  float  v8f;

// native fp32 -> bf16 (RNE) — lowers to hardware cvt, pairable by the compiler
__device__ __forceinline__ bf16 f2bf(float f) { return (bf16)f; }
__device__ __forceinline__ unsigned pack2bf(float a, float b) {
  v2bf p; p[0] = (bf16)a; p[1] = (bf16)b;
  return __builtin_bit_cast(unsigned, p);
}
__device__ __forceinline__ float sigmf(float x) { return 1.f / (1.f + expf(-x)); }

// ---------------- problem constants ----------------
#define BATCH   4096
#define SEN     32
#define HDIM    256
#define MROWS   (BATCH * SEN)       // 131072
#define HEADS   4
#define HD      64
#define PAIRS   (HEADS * BATCH)     // 16384
#define EPSN    1e-5f
#define GRU_CHUNK 32768             // M/4 rows per GRU gate-GEMM chunk

// ================== generic WMMA GEMM ==================
// C[row(C)] [N] = ACT( A[row(A)] [K] @ WT[N,K](bf16, transposed) + bias[N] )
// row(A) = aOff + z*aZ + m*aStr ; row(C) = cOff + z*cZ + m*cStr
// block tile 128x64, 8 waves, each wave 32x32 (2x2 of 16x16), K step 32 (bf16)
#define BM 128
#define BN 64
#define BK 32
#define BKP (BK + 16)   // padded LDS K-stride (48 elems = 96B, 16B-aligned rows)

template <int ACT>   // 0=none 1=relu 2=tanh 3=leaky(0.01)
__global__ __launch_bounds__(256)
void gemm_bias_act(const float* __restrict__ A, const bf16* __restrict__ WT,
                   const float* __restrict__ bias, float* __restrict__ C,
                   int K, int N,
                   int aOff, int aStr, int aZ,
                   int cOff, int cStr, int cZ,
                   long wZ, int bZ)
{
  const int z  = blockIdx.z;
  const bf16*  WTz = WT   + (size_t)z * wZ;
  const float* bz  = bias + (size_t)z * bZ;
  const int m0 = blockIdx.x * BM;
  const int n0 = blockIdx.y * BN;
  const long rowBaseA = (long)(aOff + z * aZ);

  __shared__ bf16 As [BM][BKP];   // A tile, [m][k]
  __shared__ bf16 WsT[BN][BKP];   // W tile transposed, [n][k] (contig k per lane)

  const int tid = threadIdx.x;
  const int wave = tid >> 5, lane = tid & 31;
  const int laneHalf = lane >> 4, laneMod = lane & 15;
  const int wm = wave >> 1, wn = wave & 1;

  v8f acc[2][2] = {};

  for (int k0 = 0; k0 < K; k0 += BK) {
    // ---- stage A tile: 128x32 fp32 -> bf16, float4 loads / uint2 LDS stores
#pragma unroll
    for (int j = 0; j < 4; ++j) {
      int id = tid + j * 256;            // 0..1023 float4 slots
      int r  = id >> 3;                  // 0..127
      int cq = (id & 7) * 4;             // 0,4,...,28
      long gr = rowBaseA + (long)(m0 + r) * aStr;
      float4 a4 = *(const float4*)(A + gr * (long)K + k0 + cq);
      uint2 pk;
      pk.x = pack2bf(a4.x, a4.y);
      pk.y = pack2bf(a4.z, a4.w);
      *(uint2*)(&As[r][cq]) = pk;
    }
    // ---- stage W tile: 64 cols x 32 k, from transposed [N,K] global (uint4 = 8 bf16)
    {
      int c  = tid >> 2;                 // 0..63
      int rq = (tid & 3) * 8;            // 0,8,16,24
      *(uint4*)(&WsT[c][rq]) =
          *(const uint4*)(WTz + (size_t)(n0 + c) * K + k0 + rq);
    }
    // ---- prefetch next K-step lines (global_prefetch_b8)
    if (k0 + BK < K) {
      int id = tid, r = id >> 3, cq = (id & 7) * 4;
      long gr = rowBaseA + (long)(m0 + r) * aStr;
      __builtin_prefetch((const void*)(A + gr * (long)K + k0 + BK + cq), 0, 1);
      int c = tid >> 2, rq = (tid & 3) * 8;
      __builtin_prefetch((const void*)(WTz + (size_t)(n0 + c) * K + k0 + BK + rq), 0, 1);
    }
    __syncthreads();

    v16bf af[2], bfm[2];
#pragma unroll
    for (int rt = 0; rt < 2; ++rt) {
      int row = wm * 32 + rt * 16 + laneMod;
#pragma unroll
      for (int i = 0; i < 16; ++i) {
        int kk = i + laneHalf * 16;                    // contig 16-elem run
        af[rt][i] = As[row][kk];
      }
    }
#pragma unroll
    for (int ct = 0; ct < 2; ++ct) {
      int col = wn * 32 + ct * 16 + laneMod;
#pragma unroll
      for (int i = 0; i < 16; ++i) {
        int kk = i + laneHalf * 16;                    // contig 16-elem run
        bfm[ct][i] = WsT[col][kk];
      }
    }
#pragma unroll
    for (int rt = 0; rt < 2; ++rt)
#pragma unroll
      for (int ct = 0; ct < 2; ++ct)
        acc[rt][ct] = __builtin_amdgcn_wmma_f32_16x16x32_bf16(
            false, af[rt], false, bfm[ct], (short)0, acc[rt][ct], false, false);
    __syncthreads();
  }

  // ---- epilogue (branch-free per-instantiation)
#pragma unroll
  for (int rt = 0; rt < 2; ++rt) {
#pragma unroll
    for (int ct = 0; ct < 2; ++ct) {
      int ni = n0 + wn * 32 + ct * 16 + laneMod;
      float bv = bz[ni];
#pragma unroll
      for (int r = 0; r < 8; ++r) {
        int mi = m0 + wm * 32 + rt * 16 + r + 8 * laneHalf;
        long gr = (long)(cOff + z * cZ) + (long)mi * cStr;
        float v = acc[rt][ct][r] + bv;
        if (ACT == 1)      v = fmaxf(v, 0.f);
        else if (ACT == 2) v = tanhf(v);
        else if (ACT == 3) v = (v > 0.f) ? v : 0.01f * v;
        C[gr * (long)N + ni] = v;
      }
    }
  }
}

// ================== attention: scores = Qh @ Kh^T / 8 ==================
// one wave per (head g, batch b): 32x32 output, K = 64 (2 steps)
__global__ __launch_bounds__(256)
void attn_scores(const float* __restrict__ Q, const float* __restrict__ Kb,
                 float* __restrict__ Sc)
{
  const int tid = threadIdx.x;
  const int wave = tid >> 5, lane = tid & 31;
  const int laneHalf = lane >> 4, laneMod = lane & 15;
  const int p = blockIdx.x * 8 + wave;           // 0..16383
  const int g = p >> 12, b = p & 4095;
  const float* Qp = Q  + ((size_t)b * SEN) * HDIM + g * HD;
  const float* Kp = Kb + ((size_t)b * SEN) * HDIM + g * HD;

  v8f acc[2][2] = {};
  for (int d0 = 0; d0 < HD; d0 += 32) {
    v16bf af[2], bfm[2];
#pragma unroll
    for (int rt = 0; rt < 2; ++rt) {
      int q = rt * 16 + laneMod;
#pragma unroll
      for (int i = 0; i < 16; ++i) {
        int kk = i + laneHalf * 16;
        af[rt][i] = f2bf(Qp[(size_t)q * HDIM + d0 + kk]);
      }
    }
#pragma unroll
    for (int ct = 0; ct < 2; ++ct) {
      int k = ct * 16 + laneMod;
#pragma unroll
      for (int i = 0; i < 16; ++i) {
        int kk = i + laneHalf * 16;
        bfm[ct][i] = f2bf(Kp[(size_t)k * HDIM + d0 + kk]);
      }
    }
#pragma unroll
    for (int rt = 0; rt < 2; ++rt)
#pragma unroll
      for (int ct = 0; ct < 2; ++ct)
        acc[rt][ct] = __builtin_amdgcn_wmma_f32_16x16x32_bf16(
            false, af[rt], false, bfm[ct], (short)0, acc[rt][ct], false, false);
  }
#pragma unroll
  for (int rt = 0; rt < 2; ++rt)
#pragma unroll
    for (int ct = 0; ct < 2; ++ct)
#pragma unroll
      for (int r = 0; r < 8; ++r) {
        int q = rt * 16 + r + 8 * laneHalf;
        int k = ct * 16 + laneMod;
        Sc[(size_t)p * 1024 + q * 32 + k] = acc[rt][ct][r] * 0.125f;
      }
}

// legacy softmax(axis=0): per (q,k) reduce max & sum-exp over all 16384 pairs
__global__ __launch_bounds__(256)
void softmax_stats(const float* __restrict__ Sc, float* __restrict__ smax,
                   float* __restrict__ ssum)
{
  const int qk = blockIdx.x;      // 0..1023
  const int t = threadIdx.x;
  __shared__ float sm[256];
  float m = -1e30f;
  for (int bb = t; bb < PAIRS; bb += 256) m = fmaxf(m, Sc[(size_t)bb * 1024 + qk]);
  sm[t] = m; __syncthreads();
  for (int s = 128; s > 0; s >>= 1) { if (t < s) sm[t] = fmaxf(sm[t], sm[t + s]); __syncthreads(); }
  float mx = sm[0]; __syncthreads();
  float acc = 0.f;
  for (int bb = t; bb < PAIRS; bb += 256) acc += expf(Sc[(size_t)bb * 1024 + qk] - mx);
  sm[t] = acc; __syncthreads();
  for (int s = 128; s > 0; s >>= 1) { if (t < s) sm[t] += sm[t + s]; __syncthreads(); }
  if (t == 0) { smax[qk] = mx; ssum[qk] = sm[0]; }
}

// out = relu( wts @ Vh  + Vr ), written back into X (pre-norm)
__global__ __launch_bounds__(256)
void attn_out(const float* __restrict__ Sc, const float* __restrict__ smax,
              const float* __restrict__ ssum, const float* __restrict__ V,
              const float* __restrict__ Vr, float* __restrict__ X)
{
  const int tid = threadIdx.x;
  const int wave = tid >> 5, lane = tid & 31;
  const int laneHalf = lane >> 4, laneMod = lane & 15;
  const int p = blockIdx.x * 8 + wave;
  const int g = p >> 12, b = p & 4095;

  v8f acc[2][4] = {};
  v16bf af[2];
#pragma unroll
  for (int rt = 0; rt < 2; ++rt) {
    int q = rt * 16 + laneMod;
#pragma unroll
    for (int i = 0; i < 16; ++i) {
      int kk = i + laneHalf * 16;                       // logical k (0..31)
      int qk = q * 32 + kk;
      float s = Sc[(size_t)p * 1024 + qk];
      af[rt][i] = f2bf(expf(s - smax[qk]) / ssum[qk]);
    }
  }
  v16bf bfm[4];
#pragma unroll
  for (int ct = 0; ct < 4; ++ct) {
    int d = ct * 16 + laneMod;
#pragma unroll
    for (int i = 0; i < 16; ++i) {
      int kk = i + laneHalf * 16;                       // k row of Vh
      bfm[ct][i] = f2bf(V[((size_t)b * SEN + kk) * HDIM + g * HD + d]);
    }
  }
#pragma unroll
  for (int rt = 0; rt < 2; ++rt)
#pragma unroll
    for (int ct = 0; ct < 4; ++ct)
      acc[rt][ct] = __builtin_amdgcn_wmma_f32_16x16x32_bf16(
          false, af[rt], false, bfm[ct], (short)0, acc[rt][ct], false, false);

#pragma unroll
  for (int rt = 0; rt < 2; ++rt)
#pragma unroll
    for (int ct = 0; ct < 4; ++ct)
#pragma unroll
      for (int r = 0; r < 8; ++r) {
        int q = rt * 16 + r + 8 * laneHalf;
        int d = ct * 16 + laneMod;
        size_t idx = ((size_t)b * SEN + q) * HDIM + g * HD + d;
        X[idx] = fmaxf(acc[rt][ct][r] + Vr[idx], 0.f);
      }
}

// InstanceNorm over last dim (per row) + affine
__global__ __launch_bounds__(256)
void inst_norm(float* __restrict__ X, const float* __restrict__ gamma,
               const float* __restrict__ beta)
{
  const int row = blockIdx.x, t = threadIdx.x;
  __shared__ float sm[256];
  __shared__ float s_mean;
  float v = X[(size_t)row * HDIM + t];
  sm[t] = v; __syncthreads();
  for (int s = 128; s > 0; s >>= 1) { if (t < s) sm[t] += sm[t + s]; __syncthreads(); }
  if (t == 0) s_mean = sm[0] * (1.f / HDIM);
  __syncthreads();
  float mean = s_mean;
  float d = v - mean;
  sm[t] = d * d; __syncthreads();
  for (int s = 128; s > 0; s >>= 1) { if (t < s) sm[t] += sm[t + s]; __syncthreads(); }
  float var = sm[0] * (1.f / HDIM);
  X[(size_t)row * HDIM + t] = gamma[t] * (d * rsqrtf(var + EPSN)) + beta[t];
}

// s2 = (sum over senders) - s1  (per batch b, per column h)
__global__ __launch_bounds__(256)
void senders_sum_sub(const float* __restrict__ s1, float* __restrict__ s2)
{
  const int b = blockIdx.x, h = threadIdx.x;
  float acc = 0.f;
  for (int s = 0; s < SEN; ++s) acc += s1[((size_t)b * SEN + s) * HDIM + h];
  for (int s = 0; s < SEN; ++s) {
    size_t i = ((size_t)b * SEN + s) * HDIM + h;
    s2[i] = acc - s1[i];
  }
}

// GRU gates + residual, in place on X (rows [rowBase, rowBase+chunk))
__global__ __launch_bounds__(256)
void gru_update(const float* __restrict__ gi, const float* __restrict__ gh,
                float* __restrict__ X, int rowBase)
{
  const int m = blockIdx.x;        // local row in chunk
  const int j = threadIdx.x;       // 0..255
  const float* gim = gi + (size_t)m * 768;
  const float* ghm = gh + (size_t)m * 768;
  float r = sigmf(gim[j]       + ghm[j]);
  float z = sigmf(gim[j + 256] + ghm[j + 256]);
  float n = tanhf(gim[j + 512] + r * ghm[j + 512]);
  float* xp = X + ((size_t)(rowBase + m)) * HDIM + j;
  float h = *xp;
  *xp = h + (1.f - z) * n + z * h;   // residual + GRU blend
}

// cred = sigmoid( sum(m1*m2, -1) )
__global__ __launch_bounds__(256)
void cred_kernel(const float* __restrict__ m1, const float* __restrict__ m2,
                 float* __restrict__ out)
{
  const int row = blockIdx.x, t = threadIdx.x;
  __shared__ float sm[256];
  size_t i = (size_t)row * HDIM + t;
  sm[t] = m1[i] * m2[i]; __syncthreads();
  for (int s = 128; s > 0; s >>= 1) { if (t < s) sm[t] += sm[t + s]; __syncthreads(); }
  if (t == 0) out[row] = sigmf(sm[0]);
}

// ---- weight converters ----
// elementwise fp32 -> bf16 (used when source is already [N,K])
__global__ void conv_bf(const float* __restrict__ s, bf16* __restrict__ d, int n)
{
  int i = blockIdx.x * 256 + threadIdx.x;
  if (i < n) d[i] = f2bf(s[i]);
}
// batched transpose: src z-th matrix [K,N] -> dst z-th matrix [N,K]
__global__ void conv_wT(const float* __restrict__ s, bf16* __restrict__ d,
                        int K, int N, int total)
{
  int i = blockIdx.x * 256 + threadIdx.x;
  if (i < total) {
    int z = i / (K * N);
    int rem = i - z * (K * N);
    int k = rem / N, n = rem % N;
    d[(size_t)z * K * N + (size_t)n * K + k] = f2bf(s[i]);
  }
}

// ================== host side ==================
extern "C" void kernel_launch(void* const* d_in, const int* in_sizes, int n_in,
                              void* d_out, int out_size, void* d_ws, size_t ws_size,
                              hipStream_t stream)
{
  (void)in_sizes; (void)n_in; (void)out_size; (void)ws_size;
  const float* inputs  = (const float*)d_in[0];
  const float* embed_w = (const float*)d_in[1];
  const float* embed_b = (const float*)d_in[2];
  const float* wq = (const float*)d_in[3];  const float* bq = (const float*)d_in[4];
  const float* wk = (const float*)d_in[5];  const float* bk = (const float*)d_in[6];
  const float* wv = (const float*)d_in[7];  const float* bv = (const float*)d_in[8];
  const float* wvr = (const float*)d_in[9]; const float* bvr = (const float*)d_in[10];
  const float* gamma = (const float*)d_in[11];
  const float* beta  = (const float*)d_in[12];
  const float* mp_w1 = (const float*)d_in[13]; const float* mp_b1 = (const float*)d_in[14];
  const float* mp_w3 = (const float*)d_in[15]; const float* mp_b3 = (const float*)d_in[16];
  const float* gru_wih = (const float*)d_in[17]; const float* gru_whh = (const float*)d_in[18];
  const float* gru_bih = (const float*)d_in[19]; const float* gru_bhh = (const float*)d_in[20];
  const float* mlp1_w = (const float*)d_in[21]; const float* mlp1_b = (const float*)d_in[22];
  const float* mlp2_w = (const float*)d_in[23]; const float* mlp2_b = (const float*)d_in[24];

  float* out_cred = (float*)d_out;                 // [131072]
  float* out_fs   = (float*)d_out + MROWS;         // [131072*256]

  // ---- workspace carve-out ----
  char* ws = (char*)d_ws;
  size_t off = 0;
  auto carve = [&](size_t bytes) { void* p = ws + off; off += (bytes + 255) & ~(size_t)255; return p; };
  const size_t ACT_BYTES = (size_t)MROWS * HDIM * sizeof(float);      // 134 MB
  float* XB  = (float*)carve(ACT_BYTES);   // activations / state
  float* BQ  = (float*)carve(ACT_BYTES);   // Q / s1 / gi-chunk / m1
  float* BKb = (float*)carve(ACT_BYTES);   // K / s2 / m2
  float* BV  = (float*)carve(ACT_BYTES);   // V / s3
  float* BVR = (float*)carve(ACT_BYTES);   // Vr / gh-chunk
  float* SC  = (float*)carve((size_t)PAIRS * 1024 * sizeof(float));   // 67 MB
  float* SMX = (float*)carve(1024 * sizeof(float));
  float* SSM = (float*)carve(1024 * sizeof(float));
  // bf16 weights, all stored TRANSPOSED: [N, K]
  bf16* embW  = (bf16*)carve(65536 * 2);
  bf16* wqB   = (bf16*)carve(65536 * 2);
  bf16* wkB   = (bf16*)carve(65536 * 2);
  bf16* wvB   = (bf16*)carve(65536 * 2);
  bf16* wvrB  = (bf16*)carve(65536 * 2);
  bf16* w1B   = (bf16*)carve((size_t)SEN * 65536 * 2);   // 4 MB
  bf16* w3B   = (bf16*)carve((size_t)SEN * 65536 * 2);   // 4 MB
  bf16* wihT  = (bf16*)carve((size_t)256 * 768 * 2);     // gru_wih is already [N=768,K=256]
  bf16* whhT  = (bf16*)carve((size_t)256 * 768 * 2);
  bf16* m1B   = (bf16*)carve(65536 * 2);
  bf16* m2B   = (bf16*)carve(65536 * 2);

  // ---- weight conversion (transpose [K,N] -> [N,K] where needed) ----
  conv_wT<<<256, 256, 0, stream>>>(embed_w, embW, HDIM, HDIM, 65536);
  conv_wT<<<256, 256, 0, stream>>>(wq,  wqB,  HDIM, HDIM, 65536);
  conv_wT<<<256, 256, 0, stream>>>(wk,  wkB,  HDIM, HDIM, 65536);
  conv_wT<<<256, 256, 0, stream>>>(wv,  wvB,  HDIM, HDIM, 65536);
  conv_wT<<<256, 256, 0, stream>>>(wvr, wvrB, HDIM, HDIM, 65536);
  conv_wT<<<8192, 256, 0, stream>>>(mp_w1, w1B, HDIM, HDIM, SEN * 65536);
  conv_wT<<<8192, 256, 0, stream>>>(mp_w3, w3B, HDIM, HDIM, SEN * 65536);
  conv_bf<<<768, 256, 0, stream>>>(gru_wih, wihT, 768 * 256);  // x @ wih.T : [N,K] native
  conv_bf<<<768, 256, 0, stream>>>(gru_whh, whhT, 768 * 256);
  conv_wT<<<256, 256, 0, stream>>>(mlp1_w, m1B, HDIM, HDIM, 65536);
  conv_wT<<<256, 256, 0, stream>>>(mlp2_w, m2B, HDIM, HDIM, 65536);

  const dim3 gFull(MROWS / BM, HDIM / BN, 1);     // (1024, 4)
  const dim3 blk(256);

  // ---- embed: X = tanh(inputs @ W + b) ----
  gemm_bias_act<2><<<gFull, blk, 0, stream>>>(inputs, embW, embed_b, XB,
      HDIM, HDIM, 0, 1, 0, 0, 1, 0, 0, 0);

  // ---- attention projections (relu) ----
  gemm_bias_act<1><<<gFull, blk, 0, stream>>>(XB, wqB,  bq,  BQ,  HDIM, HDIM, 0,1,0, 0,1,0, 0,0);
  gemm_bias_act<1><<<gFull, blk, 0, stream>>>(XB, wkB,  bk,  BKb, HDIM, HDIM, 0,1,0, 0,1,0, 0,0);
  gemm_bias_act<1><<<gFull, blk, 0, stream>>>(XB, wvB,  bv,  BV,  HDIM, HDIM, 0,1,0, 0,1,0, 0,0);
  gemm_bias_act<1><<<gFull, blk, 0, stream>>>(XB, wvrB, bvr, BVR, HDIM, HDIM, 0,1,0, 0,1,0, 0,0);

  // ---- attention core ----
  attn_scores<<<PAIRS / 8, blk, 0, stream>>>(BQ, BKb, SC);
  softmax_stats<<<1024, blk, 0, stream>>>(SC, SMX, SSM);
  attn_out<<<PAIRS / 8, blk, 0, stream>>>(SC, SMX, SSM, BV, BVR, XB);
  inst_norm<<<MROWS, blk, 0, stream>>>(XB, gamma, beta);

  // ---- 4 GNN iterations (fs saved after the first) ----
  const dim3 gSend(BATCH / BM, HDIM / BN, SEN);          // (32, 4, 32)
  const dim3 gGate(GRU_CHUNK / BM, 768 / BN, 1);         // (256, 12)
  for (int it = 0; it < 4; ++it) {
    // s1[b,s,:] = X[b,s,:] @ mp_w1[s] + mp_b1[s]
    gemm_bias_act<0><<<gSend, blk, 0, stream>>>(XB, w1B, mp_b1, BQ,
        HDIM, HDIM, 0, SEN, 1, 0, SEN, 1, 65536, HDIM);
    senders_sum_sub<<<BATCH, blk, 0, stream>>>(BQ, BKb);
    // s3 = s2 @ mp_w3[s] + mp_b3[s]
    gemm_bias_act<0><<<gSend, blk, 0, stream>>>(BKb, w3B, mp_b3, BV,
        HDIM, HDIM, 0, SEN, 1, 0, SEN, 1, 65536, HDIM);
    // GRU, chunked over rows to cap workspace
    for (int c = 0; c < MROWS / GRU_CHUNK; ++c) {
      int base = c * GRU_CHUNK;
      gemm_bias_act<0><<<gGate, blk, 0, stream>>>(BV, wihT, gru_bih, BQ,
          HDIM, 768, base, 1, 0, 0, 1, 0, 0, 0);
      gemm_bias_act<0><<<gGate, blk, 0, stream>>>(XB, whhT, gru_bhh, BVR,
          HDIM, 768, base, 1, 0, 0, 1, 0, 0, 0);
      gru_update<<<GRU_CHUNK, blk, 0, stream>>>(BQ, BVR, XB, base);
    }
    if (it == 0) {
      hipMemcpyAsync(out_fs, XB, ACT_BYTES, hipMemcpyDeviceToDevice, stream);
    }
  }

  // ---- final MLP + credibility ----
  gemm_bias_act<2><<<gFull, blk, 0, stream>>>(XB, m1B, mlp1_b, BQ,  HDIM, HDIM, 0,1,0, 0,1,0, 0,0);
  gemm_bias_act<3><<<gFull, blk, 0, stream>>>(XB, m2B, mlp2_b, BKb, HDIM, HDIM, 0,1,0, 0,1,0, 0,0);
  cred_kernel<<<MROWS, blk, 0, stream>>>(BQ, BKb, out_cred);
}
